// DenoisingLatentEdgeNetwork_38199439130849
// MI455X (gfx1250) — compile-verified, matrix operand order checked by reading, not run on hardware
//
#include <hip/hip_runtime.h>
#include <hip/hip_bf16.h>

// ---------------- problem constants ----------------
enum : int {
  N = 2048, E = 65536, G = 32,
  F_ATOM = 16, LAT = 64, IN = 128, ED = 32, NB = 5, L = 5,
  KMSG = 2 * IN + ED,   // 288
  KNODE = 2 * IN        // 256
};

// ---------------- WMMA types / helpers ----------------
typedef __attribute__((ext_vector_type(16))) __bf16 bf16x16;
typedef __attribute__((ext_vector_type(8)))  float  v8f;

struct alignas(16) US8  { unsigned long long x, y; };          // 16 bytes
struct alignas(32) US16 { US8 lo, hi; };                       // 32 bytes

__device__ __forceinline__ unsigned short f2bf(float x) {
  unsigned int u = __builtin_bit_cast(unsigned int, x);
  u += 0x7FFFu + ((u >> 16) & 1u);          // round-to-nearest-even
  return (unsigned short)(u >> 16);
}
__device__ __forceinline__ float silu(float x) {
  return x * (1.0f / (1.0f + __expf(-x)));
}

// A fragment: base points at lds_row + k0 + 8*hi (16B aligned); second half at +16 ushorts
__device__ __forceinline__ bf16x16 load_a_frag(const unsigned short* base) {
  US16 r; r.lo = *(const US8*)base; r.hi = *(const US8*)(base + 16);
  return __builtin_bit_cast(bf16x16, r);
}
// B fragment: 16 contiguous bf16 per lane in swizzled weight buffer
__device__ __forceinline__ bf16x16 load_b_frag(const unsigned short* p) {
  US16 r; r.lo = *(const US8*)p; r.hi = *(const US8*)(p + 8);
  return __builtin_bit_cast(bf16x16, r);
}
#define WMMA_BF16(acc, a, b) \
  (acc) = __builtin_amdgcn_wmma_f32_16x16x32_bf16(false, (a), false, (b), (short)0, (acc), false, false)

__device__ __forceinline__ void atomicAddF(float* p, float v) {
  unsafeAtomicAdd(p, v);   // global_atomic_add_f32 on gfx1250
}

// ---------------- tiny setup kernels ----------------
__global__ void k_zero_f(float* __restrict__ p, int n) {
  int i = blockIdx.x * blockDim.x + threadIdx.x;
  if (i < n) p[i] = 0.0f;
}

__global__ void k_temb(const float* __restrict__ t,
                       const float* __restrict__ Wt_a, const float* __restrict__ bt_a,
                       const float* __restrict__ Wt_b, const float* __restrict__ bt_b,
                       float* __restrict__ ta, float* __restrict__ tb) {
  int tid = blockIdx.x * blockDim.x + threadIdx.x;
  if (tid >= G * (IN + ED)) return;
  int g = tid / (IN + ED), c = tid % (IN + ED);
  float tv = t[g];
  if (c < IN) ta[g * IN + c] = tv * Wt_a[c] + bt_a[c];
  else { int cc = c - IN; tb[g * ED + cc] = tv * Wt_b[cc] + bt_b[cc]; }
}

// s0pt = x @ W_atom + b_atom + ta[batch]
__global__ void k_sinit_a(const float* __restrict__ x, const float* __restrict__ W_atom,
                          const float* __restrict__ b_atom, const int* __restrict__ batch,
                          const float* __restrict__ ta, float* __restrict__ s0pt) {
  int tid = blockIdx.x * blockDim.x + threadIdx.x;
  if (tid >= N * IN) return;
  int n = tid / IN, c = tid % IN;
  float acc = b_atom[c] + ta[batch[n] * IN + c];
  #pragma unroll
  for (int k = 0; k < F_ATOM; ++k) acc += x[n * F_ATOM + k] * W_atom[k * IN + c];
  s0pt[tid] = acc;
}

// s = s0pt @ W_at + b_at + z @ W_lat + b_lat  (one 128-thread block per row)
__global__ __launch_bounds__(IN) void k_sinit_b(const float* __restrict__ s0pt,
                                                const float* __restrict__ W_at, const float* __restrict__ b_at,
                                                const float* __restrict__ z, const float* __restrict__ W_lat,
                                                const float* __restrict__ b_lat,
                                                float* __restrict__ s, unsigned short* __restrict__ s_bf) {
  __shared__ float r0[IN]; __shared__ float rz[LAT];
  int n = blockIdx.x, c = threadIdx.x;
  r0[c] = s0pt[n * IN + c];
  if (c < LAT) rz[c] = z[n * LAT + c];
  __syncthreads();
  float acc = b_at[c] + b_lat[c];
  for (int k = 0; k < IN; ++k)  acc += r0[k] * W_at[k * IN + c];
  for (int k = 0; k < LAT; ++k) acc += rz[k] * W_lat[k * IN + c];
  s[n * IN + c] = acc;
  s_bf[n * IN + c] = f2bf(acc);
}

// etmp = edge_attr @ W_bond + b_bond + tb[beg]
__global__ void k_einit_a(const float* __restrict__ edge_attr, const float* __restrict__ W_bond,
                          const float* __restrict__ b_bond, const int* __restrict__ beg,
                          const float* __restrict__ tb, float* __restrict__ etmp) {
  int tid = blockIdx.x * blockDim.x + threadIdx.x;
  if (tid >= E * ED) return;
  int k = tid / ED, c = tid % ED;
  float acc = b_bond[c] + tb[beg[k] * ED + c];
  #pragma unroll
  for (int q = 0; q < NB; ++q) acc += edge_attr[k * NB + q] * W_bond[q * ED + c];
  etmp[tid] = acc;
}

// e = etmp @ W_bt + b_bt  (8 edges per 256-thread block)
__global__ __launch_bounds__(256) void k_einit_b(const float* __restrict__ etmp,
                                                 const float* __restrict__ W_bt, const float* __restrict__ b_bt,
                                                 float* __restrict__ e, unsigned short* __restrict__ e_bf) {
  __shared__ float rows[8][ED];
  int e0 = blockIdx.x * 8;
  int lr = threadIdx.x / ED, c = threadIdx.x % ED;
  rows[lr][c] = etmp[(size_t)(e0 + lr) * ED + c];
  __syncthreads();
  float acc = b_bt[c];
  for (int q = 0; q < ED; ++q) acc += rows[lr][q] * W_bt[q * ED + c];
  size_t idx = (size_t)(e0 + lr) * ED + c;
  e[idx] = acc; e_bf[idx] = f2bf(acc);
}

// Convert + swizzle fp32 weights [K][Ncols] into per-lane WMMA B-fragment order:
// out[((kb*numCT+ct)*32 + lane)*16 + el]
__global__ void k_swz(const float* __restrict__ W, unsigned short* __restrict__ out,
                      int K, int Ncols) {
  int tid = blockIdx.x * blockDim.x + threadIdx.x;
  int total = K * Ncols;
  if (tid >= total) return;
  int numCT = Ncols >> 4;
  int el = tid & 15;
  int lane = (tid >> 4) & 31;
  int ct = (tid >> 9) % numCT;
  int kb = tid / (512 * numCT);
  int hi = lane >> 4, n = ct * 16 + (lane & 15);
  int k = kb * 32 + (el < 8 ? 8 * hi + el : 16 + 8 * hi + (el - 8));
  out[tid] = f2bf(W[(size_t)k * Ncols + n]);
}

// ---------------- main WMMA kernels ----------------
#define LDC 296   // 288 + 8 pad (592B rows, 16B aligned)
#define LDN 264   // 256 + 8 pad

// messages: m = silu([s[i],s[j],e] @ Wmsg + b); agg[i] += m   (16 edges / block, 4 waves)
__global__ __launch_bounds__(128) void k_msg(const unsigned short* __restrict__ s_bf,
                                             const unsigned short* __restrict__ e_bf,
                                             const int* __restrict__ edge_index,
                                             const unsigned short* __restrict__ Wsw,
                                             const float* __restrict__ bias,
                                             float* __restrict__ agg) {
  __shared__ alignas(16) unsigned short cat[16 * LDC];
  __shared__ int iIdx[16], jIdx[16];
  int tid = threadIdx.x;
  int edge0 = blockIdx.x * 16;
  if (tid < 16) {
    jIdx[tid] = edge_index[edge0 + tid];
    iIdx[tid] = edge_index[E + edge0 + tid];
  }
  __syncthreads();
  for (int q = tid; q < 16 * 36; q += 128) {
    int row = q / 36; int col0 = (q % 36) * 8;
    const unsigned short* src;
    if (col0 < IN)            src = s_bf + (size_t)iIdx[row] * IN + col0;
    else if (col0 < 2 * IN)   src = s_bf + (size_t)jIdx[row] * IN + (col0 - IN);
    else                      src = e_bf + (size_t)(edge0 + row) * ED + (col0 - 2 * IN);
    *(US8*)&cat[row * LDC + col0] = *(const US8*)src;
  }
  __syncthreads();
  int wave = tid >> 5, lane = tid & 31;
  int m = lane & 15, hi = lane >> 4, n = lane & 15;
  v8f acc0 = {}; v8f acc1 = {};
  int ct0 = wave, ct1 = wave + 4;
  #pragma unroll
  for (int kb = 0; kb < KMSG / 32; ++kb) {
    bf16x16 a  = load_a_frag(&cat[m * LDC + kb * 32 + 8 * hi]);
    bf16x16 b0 = load_b_frag(Wsw + (size_t)((kb * 8 + ct0) * 32 + lane) * 16);
    WMMA_BF16(acc0, a, b0);
    bf16x16 b1 = load_b_frag(Wsw + (size_t)((kb * 8 + ct1) * 32 + lane) * 16);
    WMMA_BF16(acc1, a, b1);
  }
  #pragma unroll
  for (int r = 0; r < 8; ++r) {
    int orow = r + 8 * hi;
    int node = iIdx[orow];
    int c0 = ct0 * 16 + n, c1 = ct1 * 16 + n;
    atomicAddF(&agg[(size_t)node * IN + c0], silu(acc0[r] + bias[c0]));
    atomicAddF(&agg[(size_t)node * IN + c1], silu(acc1[r] + bias[c1]));
  }
}

// node update: s += silu([s,agg] @ Wnode + b)   (16 nodes / block, 4 waves)
__global__ __launch_bounds__(128) void k_node(const float* __restrict__ agg,
                                              const unsigned short* __restrict__ Wsw,
                                              const float* __restrict__ bias,
                                              float* __restrict__ s,
                                              unsigned short* __restrict__ s_bf) {
  __shared__ alignas(16) unsigned short cat[16 * LDN];
  int tid = threadIdx.x;
  int node0 = blockIdx.x * 16;
  for (int q = tid; q < 16 * 32; q += 128) {
    int row = q >> 5; int col0 = (q & 31) * 8;
    unsigned short* dst = &cat[row * LDN + col0];
    if (col0 < IN) {
      *(US8*)dst = *(const US8*)(s_bf + (size_t)(node0 + row) * IN + col0);
    } else {
      const float* srcf = agg + (size_t)(node0 + row) * IN + (col0 - IN);
      #pragma unroll
      for (int t = 0; t < 8; ++t) dst[t] = f2bf(srcf[t]);
    }
  }
  __syncthreads();
  int wave = tid >> 5, lane = tid & 31;
  int m = lane & 15, hi = lane >> 4, n = lane & 15;
  v8f acc0 = {}; v8f acc1 = {};
  int ct0 = wave, ct1 = wave + 4;
  #pragma unroll
  for (int kb = 0; kb < KNODE / 32; ++kb) {
    bf16x16 a  = load_a_frag(&cat[m * LDN + kb * 32 + 8 * hi]);
    bf16x16 b0 = load_b_frag(Wsw + (size_t)((kb * 8 + ct0) * 32 + lane) * 16);
    WMMA_BF16(acc0, a, b0);
    bf16x16 b1 = load_b_frag(Wsw + (size_t)((kb * 8 + ct1) * 32 + lane) * 16);
    WMMA_BF16(acc1, a, b1);
  }
  #pragma unroll
  for (int r = 0; r < 8; ++r) {
    int node = node0 + r + 8 * hi;
    int c0 = ct0 * 16 + n, c1 = ct1 * 16 + n;
    float sv0 = s[(size_t)node * IN + c0] + silu(acc0[r] + bias[c0]);
    float sv1 = s[(size_t)node * IN + c1] + silu(acc1[r] + bias[c1]);
    s[(size_t)node * IN + c0] = sv0; s_bf[(size_t)node * IN + c0] = f2bf(sv0);
    s[(size_t)node * IN + c1] = sv1; s_bf[(size_t)node * IN + c1] = f2bf(sv1);
  }
}

// edge update: e += silu([s[i],s[j],e] @ Wedge + b)   (16 edges / block, 2 waves)
__global__ __launch_bounds__(64) void k_edge(const unsigned short* __restrict__ s_bf,
                                             const unsigned short* __restrict__ e_bf,
                                             const int* __restrict__ edge_index,
                                             const unsigned short* __restrict__ Wsw,
                                             const float* __restrict__ bias,
                                             float* __restrict__ e,
                                             unsigned short* __restrict__ e_bf_out) {
  __shared__ alignas(16) unsigned short cat[16 * LDC];
  __shared__ int iIdx[16], jIdx[16];
  int tid = threadIdx.x;
  int edge0 = blockIdx.x * 16;
  if (tid < 16) {
    jIdx[tid] = edge_index[edge0 + tid];
    iIdx[tid] = edge_index[E + edge0 + tid];
  }
  __syncthreads();
  for (int q = tid; q < 16 * 36; q += 64) {
    int row = q / 36; int col0 = (q % 36) * 8;
    const unsigned short* src;
    if (col0 < IN)            src = s_bf + (size_t)iIdx[row] * IN + col0;
    else if (col0 < 2 * IN)   src = s_bf + (size_t)jIdx[row] * IN + (col0 - IN);
    else                      src = e_bf + (size_t)(edge0 + row) * ED + (col0 - 2 * IN);
    *(US8*)&cat[row * LDC + col0] = *(const US8*)src;
  }
  __syncthreads();
  int wave = tid >> 5, lane = tid & 31;
  int m = lane & 15, hi = lane >> 4, n = lane & 15;
  v8f acc = {};
  int ct = wave;           // numCT = 2
  #pragma unroll
  for (int kb = 0; kb < KMSG / 32; ++kb) {
    bf16x16 a = load_a_frag(&cat[m * LDC + kb * 32 + 8 * hi]);
    bf16x16 b = load_b_frag(Wsw + (size_t)((kb * 2 + ct) * 32 + lane) * 16);
    WMMA_BF16(acc, a, b);
  }
  #pragma unroll
  for (int r = 0; r < 8; ++r) {
    int orow = r + 8 * hi;
    size_t idx = (size_t)(edge0 + orow) * ED + ct * 16 + n;
    float ev = e[idx] + silu(acc[r] + bias[ct * 16 + n]);
    e[idx] = ev; e_bf_out[idx] = f2bf(ev);
  }
}

// ---------------- readout kernels ----------------
__global__ __launch_bounds__(IN) void k_s2(const float* __restrict__ s, const float* __restrict__ Wsh,
                                           const float* __restrict__ bsh, float* __restrict__ s2) {
  __shared__ float row[IN];
  int n = blockIdx.x, c = threadIdx.x;
  row[c] = s[(size_t)n * IN + c];
  __syncthreads();
  float acc = bsh[c];
  for (int k = 0; k < IN; ++k) acc += row[k] * Wsh[k * IN + c];
  s2[(size_t)n * IN + c] = silu(acc);
}

__global__ __launch_bounds__(IN) void k_al(const float* __restrict__ s2, const float* __restrict__ Wal,
                                           const float* __restrict__ bal,
                                           float* __restrict__ out_latent, float* __restrict__ out_atoms) {
  __shared__ float row[IN];
  int n = blockIdx.x, c = threadIdx.x;
  row[c] = s2[(size_t)n * IN + c];
  __syncthreads();
  if (c < F_ATOM + LAT) {
    float acc = bal[c];
    for (int k = 0; k < IN; ++k) acc += row[k] * Wal[k * (F_ATOM + LAT) + c];
    if (c < F_ATOM) out_atoms[(size_t)n * F_ATOM + c] = acc;
    else            out_latent[(size_t)n * LAT + (c - F_ATOM)] = acc;
  }
}

__global__ void k_scatter_idx(const int* __restrict__ edge_index, int* __restrict__ idx_map) {
  int k = blockIdx.x * blockDim.x + threadIdx.x;
  if (k >= E) return;
  int j = edge_index[k], i = edge_index[E + k];
  atomicMax(&idx_map[(size_t)j * N + i], k + 1);
}

__global__ void k_esym(const int* __restrict__ edge_index, const int* __restrict__ idx_map,
                       const float* __restrict__ e, unsigned short* __restrict__ esym_bf) {
  int tid = blockIdx.x * blockDim.x + threadIdx.x;
  if (tid >= E * ED) return;
  int k = tid / ED, c = tid % ED;
  int j = edge_index[k], i = edge_index[E + k];
  int f = idx_map[(size_t)j * N + i] - 1;
  int r = idx_map[(size_t)i * N + j] - 1;
  float vf = (f >= 0) ? e[(size_t)f * ED + c] : 0.0f;
  float vr = (r >= 0) ? e[(size_t)r * ED + c] : 0.0f;
  esym_bf[tid] = f2bf(0.5f * (vf + vr));
}

// f = silu(s2[i]+s2[j]+esym@Wbm+bbm); bonds = f@Wbl+bbl   (16 edges/block, 4 waves)
__global__ __launch_bounds__(128) void k_bond(const unsigned short* __restrict__ esym_bf,
                                              const float* __restrict__ s2,
                                              const int* __restrict__ edge_index,
                                              const unsigned short* __restrict__ Wbm_sw,
                                              const float* __restrict__ bbm,
                                              const float* __restrict__ Wbl,
                                              const float* __restrict__ bbl,
                                              float* __restrict__ bonds_out) {
  __shared__ alignas(16) unsigned short atile[16 * ED];
  __shared__ float fls[16 * IN];
  __shared__ int iIdx[16], jIdx[16];
  int tid = threadIdx.x;
  int edge0 = blockIdx.x * 16;
  if (tid < 16) {
    jIdx[tid] = edge_index[edge0 + tid];
    iIdx[tid] = edge_index[E + edge0 + tid];
  }
  for (int q = tid; q < 64; q += 128) {
    int row = q / 4; int col0 = (q % 4) * 8;
    *(US8*)&atile[row * ED + col0] = *(const US8*)(esym_bf + (size_t)(edge0 + row) * ED + col0);
  }
  __syncthreads();
  int wave = tid >> 5, lane = tid & 31;
  int m = lane & 15, hi = lane >> 4, n = lane & 15;
  v8f acc0 = {}; v8f acc1 = {};
  int ct0 = wave, ct1 = wave + 4;          // numCT = 8, single K-block
  bf16x16 a  = load_a_frag(&atile[m * ED + 8 * hi]);
  bf16x16 b0 = load_b_frag(Wbm_sw + (size_t)(ct0 * 32 + lane) * 16);
  WMMA_BF16(acc0, a, b0);
  bf16x16 b1 = load_b_frag(Wbm_sw + (size_t)(ct1 * 32 + lane) * 16);
  WMMA_BF16(acc1, a, b1);
  #pragma unroll
  for (int r = 0; r < 8; ++r) {
    int orow = r + 8 * hi;
    int ii = iIdx[orow], jj = jIdx[orow];
    int c0 = ct0 * 16 + n, c1 = ct1 * 16 + n;
    fls[orow * IN + c0] = silu(acc0[r] + bbm[c0] + s2[(size_t)ii * IN + c0] + s2[(size_t)jj * IN + c0]);
    fls[orow * IN + c1] = silu(acc1[r] + bbm[c1] + s2[(size_t)ii * IN + c1] + s2[(size_t)jj * IN + c1]);
  }
  __syncthreads();
  if (tid < 16 * NB) {
    int row = tid / NB, c = tid % NB;
    float acc = bbl[c];
    for (int k = 0; k < IN; ++k) acc += fls[row * IN + k] * Wbl[k * NB + c];
    bonds_out[(size_t)(edge0 + row) * NB + c] = acc;
  }
}

// ---------------- host launcher ----------------
extern "C" void kernel_launch(void* const* d_in, const int* in_sizes, int n_in,
                              void* d_out, int out_size, void* d_ws, size_t ws_size,
                              hipStream_t stream) {
  (void)in_sizes; (void)n_in; (void)out_size; (void)ws_size;

  const float* x          = (const float*)d_in[0];
  const float* t          = (const float*)d_in[1];
  const float* z          = (const float*)d_in[2];
  const int*   edge_index = (const int*)d_in[3];
  const float* edge_attr  = (const float*)d_in[4];
  const int*   batch      = (const int*)d_in[5];
  const int*   beg        = (const int*)d_in[6];
  const float* Wt_a = (const float*)d_in[7];  const float* bt_a = (const float*)d_in[8];
  const float* Wt_b = (const float*)d_in[9];  const float* bt_b = (const float*)d_in[10];
  const float* W_atom = (const float*)d_in[11]; const float* b_atom = (const float*)d_in[12];
  const float* W_bond = (const float*)d_in[13]; const float* b_bond = (const float*)d_in[14];
  const float* W_lat  = (const float*)d_in[15]; const float* b_lat  = (const float*)d_in[16];
  const float* W_at   = (const float*)d_in[17]; const float* b_at   = (const float*)d_in[18];
  const float* W_bt   = (const float*)d_in[19]; const float* b_bt   = (const float*)d_in[20];
  const float* Wmsg   = (const float*)d_in[21]; const float* bmsg   = (const float*)d_in[22];
  const float* Wnode  = (const float*)d_in[23]; const float* bnode  = (const float*)d_in[24];
  const float* Wedge  = (const float*)d_in[25]; const float* bedge  = (const float*)d_in[26];
  const float* Wsh    = (const float*)d_in[27]; const float* bsh    = (const float*)d_in[28];
  const float* Wbm    = (const float*)d_in[29]; const float* bbm    = (const float*)d_in[30];
  const float* Wbl    = (const float*)d_in[31]; const float* bbl    = (const float*)d_in[32];
  const float* Wal    = (const float*)d_in[33]; const float* bal    = (const float*)d_in[34];

  float* out_latent = (float*)d_out;                      // N*LAT
  float* out_atoms  = out_latent + (size_t)N * LAT;       // N*F_ATOM
  float* out_bonds  = out_atoms + (size_t)N * F_ATOM;     // E*NB

  // workspace carve-out
  char* ws = (char*)d_ws;
  size_t off = 0;
  auto alloc = [&](size_t bytes) -> void* {
    void* p = ws + off;
    off = (off + bytes + 255) & ~(size_t)255;
    return p;
  };
  float* s        = (float*)alloc((size_t)N * IN * 4);
  float* agg      = (float*)alloc((size_t)N * IN * 4);
  float* s2       = (float*)alloc((size_t)N * IN * 4);
  float* s0pt     = (float*)alloc((size_t)N * IN * 4);
  float* e        = (float*)alloc((size_t)E * ED * 4);
  float* etmp     = (float*)alloc((size_t)E * ED * 4);
  float* ta       = (float*)alloc((size_t)G * IN * 4);
  float* tb       = (float*)alloc((size_t)G * ED * 4);
  unsigned short* s_bf    = (unsigned short*)alloc((size_t)N * IN * 2);
  unsigned short* e_bf    = (unsigned short*)alloc((size_t)E * ED * 2);
  unsigned short* esym_bf = (unsigned short*)alloc((size_t)E * ED * 2);
  int*            idx_map = (int*)alloc((size_t)N * N * 4);
  unsigned short* Wmsg_sw  = (unsigned short*)alloc((size_t)L * KMSG * IN * 2);
  unsigned short* Wnode_sw = (unsigned short*)alloc((size_t)L * KNODE * IN * 2);
  unsigned short* Wedge_sw = (unsigned short*)alloc((size_t)L * KMSG * ED * 2);
  unsigned short* Wbm_sw   = (unsigned short*)alloc((size_t)ED * IN * 2);

  // ---- weight swizzle (bf16, fragment order) ----
  for (int l = 0; l < L; ++l) {
    k_swz<<<(KMSG * IN + 255) / 256, 256, 0, stream>>>(Wmsg + (size_t)l * KMSG * IN,
                                                       Wmsg_sw + (size_t)l * KMSG * IN, KMSG, IN);
    k_swz<<<(KNODE * IN + 255) / 256, 256, 0, stream>>>(Wnode + (size_t)l * KNODE * IN,
                                                        Wnode_sw + (size_t)l * KNODE * IN, KNODE, IN);
    k_swz<<<(KMSG * ED + 255) / 256, 256, 0, stream>>>(Wedge + (size_t)l * KMSG * ED,
                                                       Wedge_sw + (size_t)l * KMSG * ED, KMSG, ED);
  }
  k_swz<<<(ED * IN + 255) / 256, 256, 0, stream>>>(Wbm, Wbm_sw, ED, IN);

  // ---- embeddings / initial projections ----
  k_temb<<<(G * (IN + ED) + 255) / 256, 256, 0, stream>>>(t, Wt_a, bt_a, Wt_b, bt_b, ta, tb);
  k_sinit_a<<<(N * IN + 255) / 256, 256, 0, stream>>>(x, W_atom, b_atom, batch, ta, s0pt);
  k_sinit_b<<<N, IN, 0, stream>>>(s0pt, W_at, b_at, z, W_lat, b_lat, s, s_bf);
  k_einit_a<<<(E * ED + 255) / 256, 256, 0, stream>>>(edge_attr, W_bond, b_bond, beg, tb, etmp);
  k_einit_b<<<E / 8, 256, 0, stream>>>(etmp, W_bt, b_bt, e, e_bf);

  // ---- message-passing layers ----
  for (int l = 0; l < L; ++l) {
    k_zero_f<<<(N * IN + 255) / 256, 256, 0, stream>>>(agg, N * IN);
    k_msg<<<E / 16, 128, 0, stream>>>(s_bf, e_bf, edge_index,
                                      Wmsg_sw + (size_t)l * KMSG * IN, bmsg + (size_t)l * IN, agg);
    k_node<<<N / 16, 128, 0, stream>>>(agg, Wnode_sw + (size_t)l * KNODE * IN,
                                       bnode + (size_t)l * IN, s, s_bf);
    k_edge<<<E / 16, 64, 0, stream>>>(s_bf, e_bf, edge_index,
                                      Wedge_sw + (size_t)l * KMSG * ED, bedge + (size_t)l * ED, e, e_bf);
  }

  // ---- readout ----
  k_s2<<<N, IN, 0, stream>>>(s, Wsh, bsh, s2);
  k_al<<<N, IN, 0, stream>>>(s2, Wal, bal, out_latent, out_atoms);

  // ---- bond branch ----
  k_zero_f<<<((size_t)N * N + 255) / 256, 256, 0, stream>>>((float*)idx_map, N * N);
  k_scatter_idx<<<E / 256, 256, 0, stream>>>(edge_index, idx_map);
  k_esym<<<(E * ED + 255) / 256, 256, 0, stream>>>(edge_index, idx_map, e, esym_bf);
  k_bond<<<E / 16, 128, 0, stream>>>(esym_bf, s2, edge_index, Wbm_sw, bbm, Wbl, bbl, out_bonds);
}
